// FiLM_79053168050938
// MI455X (gfx1250) — matrix operand order repaired and checked by reading, not compiled, not run on hardware
//
#include <hip/hip_runtime.h>
#include <hip/hip_bf16.h>

typedef __attribute__((ext_vector_type(16))) _Float16 v16h;
typedef __attribute__((ext_vector_type(8)))  _Float16 v8h;
typedef __attribute__((ext_vector_type(8)))  float    v8f;

#define N_NODES 50000
#define N_EDGES 800000

// ---------------------------------------------------------------------------
// fp32 -> fp16 elementwise convert (node features, once per layer)
// ---------------------------------------------------------------------------
__global__ void cvt_f32_f16(const float* __restrict__ x, _Float16* __restrict__ y, int n) {
    int i = blockIdx.x * blockDim.x + threadIdx.x;
    if (i < n) y[i] = (_Float16)x[i];
}

// ---------------------------------------------------------------------------
// Pack a row-major fp32 weight matrix W (K x N) into this GEMM's B-fragment
// layout for v_wmma_f32_16x16x32_f16:
//   lane l holds column (tn*16 + (l&15)); halves t=0..15 are K = ks*32 +
//   (l>>4)*16 + t (contiguous K per lane -> the GEMM loads 2x b128 per lane).
// Wp index: ((ks*tilesN + tn)*32 + lane)*16 + t   (total K*N halves)
// ---------------------------------------------------------------------------
__global__ void pack_w(const float* __restrict__ W, _Float16* __restrict__ Wp, int K, int N) {
    int t = blockIdx.x * blockDim.x + threadIdx.x;
    if (t >= K * N) return;
    int tilesN = N >> 4;
    int idx  = t & 15;
    int lane = (t >> 4) & 31;
    int tile = t >> 9;                 // ks*tilesN + tn
    int tn = tile % tilesN;
    int ks = tile / tilesN;
    int grp = lane >> 4;
    int col = tn * 16 + (lane & 15);
    int k   = ks * 32 + grp * 16 + idx;
    Wp[t] = (_Float16)W[(size_t)k * N + col];
}

// ---------------------------------------------------------------------------
// WMMA GEMM: C(MxN,f32) = A(MxK,f16 row-major) * Wp(packed f16) [+bias]
// One wave per 16x16 tile, 8 waves per 256-thread block.
// MODE 0: C = acc (+bias[col] broadcast into accumulator init)
// MODE 1: C = relu(aux[row, N+col] * acc + aux[row, col])   (aux stride 2N)
//         -> fused FiLM-skip epilogue (out = relu(gamma_s * (x@skip_w) + beta_s))
// ---------------------------------------------------------------------------
template<int MODE>
__global__ void gemm_wmma_f16(const _Float16* __restrict__ A,
                              const _Float16* __restrict__ Wp,
                              float* __restrict__ C,
                              const float* __restrict__ bias,
                              const float* __restrict__ aux,
                              int N, int K)
{
    const int lane = threadIdx.x & 31;
    const int wave = threadIdx.x >> 5;
    const int tilesN = N >> 4;
    const int tn = blockIdx.x * 8 + wave;
    if (tn >= tilesN) return;                    // uniform per wave: EXEC stays all-1
    const int tm   = blockIdx.y;
    const int l15  = lane & 15;
    const int grp  = lane >> 4;
    const int row  = tm * 16 + l15;
    const int col  = tn * 16 + l15;

    v8f acc = {};
    if (MODE == 0) {
        if (bias) {
            float bv = bias[col];
            acc = (v8f){bv, bv, bv, bv, bv, bv, bv, bv};
        }
    }

    const _Float16* Ap = A + (size_t)row * K;
    const _Float16* Bp = Wp + ((size_t)tn * 32 + lane) * 16;
    const size_t bstride = (size_t)tilesN * 512; // halves per K-step
    const int ksteps = K >> 5;

    for (int ks = 0; ks < ksteps; ++ks) {
        // A 16x32 f16 fragment: per lane two contiguous 16B chunks
        v8h alo = *(const v8h*)(Ap + ks * 32 + grp * 16);
        v8h ahi = *(const v8h*)(Ap + ks * 32 + grp * 16 + 8);
        // B 32x16 f16 fragment: packed contiguous per lane
        v8h blo = *(const v8h*)(Bp + ks * bstride);
        v8h bhi = *(const v8h*)(Bp + ks * bstride + 8);
        v16h av = __builtin_shufflevector(alo, ahi, 0,1,2,3,4,5,6,7,8,9,10,11,12,13,14,15);
        v16h bv = __builtin_shufflevector(blo, bhi, 0,1,2,3,4,5,6,7,8,9,10,11,12,13,14,15);
        acc = __builtin_amdgcn_wmma_f32_16x16x32_f16(
            /*neg_a=*/false, av, /*neg_b=*/false, bv,
            /*c_mod=*/(short)0, acc, /*reuse_a=*/false, /*reuse_b=*/false);
    }

    // C layout: VGPR r, lane l -> M = r + 8*grp, Ncol = l&15
    float* Crow = C + (size_t)(tm * 16 + grp * 8) * N + col;
    if (MODE == 0) {
        #pragma unroll
        for (int r = 0; r < 8; ++r) Crow[(size_t)r * N] = acc[r];
    } else {
        #pragma unroll
        for (int r = 0; r < 8; ++r) {
            size_t arow = (size_t)(tm * 16 + grp * 8 + r) * (2 * N);
            float g = aux[arow + N + col];
            float b = aux[arow + col];
            Crow[(size_t)r * N] = fmaxf(g * acc[r] + b, 0.f);
        }
    }
}

// ---------------------------------------------------------------------------
// Edge stage: msg = relu(gamma[dst]*h[src] + beta[dst]); atomic mean-agg by dst
// L2-resident random access (h/bg2/agg all fit in 192MB L2).
// ---------------------------------------------------------------------------
__global__ void edge_kernel(const int* __restrict__ ei,
                            const float* __restrict__ h,
                            const float* __restrict__ bg2,
                            float* __restrict__ agg,
                            float* __restrict__ cnt,
                            int dout, int shift, long long total)
{
    long long i = (long long)blockIdx.x * blockDim.x + threadIdx.x;
    if (i >= total) return;
    int e = (int)(i >> shift);
    int c = (int)(i & (dout - 1));
    int src = ei[e];
    int dst = ei[N_EDGES + e];
    size_t brow = (size_t)dst * (2 * dout);
    float g  = bg2[brow + dout + c];
    float b  = bg2[brow + c];
    float hv = h[(size_t)src * dout + c];
    float m  = fmaxf(g * hv + b, 0.f);
    atomicAdd(agg + (size_t)dst * dout + c, m);
    if (c == 0) atomicAdd(cnt + dst, 1.0f);
}

// ---------------------------------------------------------------------------
// Finalize: B = relu(B + agg / max(cnt,1))   (layer output incl. post-relu)
// ---------------------------------------------------------------------------
__global__ void finalize_kernel(float* __restrict__ B,
                                const float* __restrict__ agg,
                                const float* __restrict__ cnt,
                                int n, int shift)
{
    int i = blockIdx.x * blockDim.x + threadIdx.x;
    if (i >= n) return;
    int row = i >> shift;
    float c = cnt[row];
    float v = B[i] + agg[i] / fmaxf(c, 1.0f);
    B[i] = fmaxf(v, 0.0f);
}

// ---------------------------------------------------------------------------
// Row-wise log-softmax over 64 cols: one wave per row, 2 cols per lane.
// ---------------------------------------------------------------------------
__global__ void logsoftmax64(const float* __restrict__ X, float* __restrict__ Y, int nrows)
{
    int row  = blockIdx.x * (blockDim.x >> 5) + (threadIdx.x >> 5);
    int lane = threadIdx.x & 31;
    if (row >= nrows) return;
    const float* xr = X + (size_t)row * 64;
    float a = xr[lane], b = xr[lane + 32];
    float m = fmaxf(a, b);
    #pragma unroll
    for (int off = 16; off > 0; off >>= 1) m = fmaxf(m, __shfl_xor(m, off, 32));
    float s = __expf(a - m) + __expf(b - m);
    #pragma unroll
    for (int off = 16; off > 0; off >>= 1) s += __shfl_xor(s, off, 32);
    float lse = m + __logf(s);
    Y[(size_t)row * 64 + lane]      = a - lse;
    Y[(size_t)row * 64 + lane + 32] = b - lse;
}

// ---------------------------------------------------------------------------
extern "C" void kernel_launch(void* const* d_in, const int* in_sizes, int n_in,
                              void* d_out, int out_size, void* d_ws, size_t ws_size,
                              hipStream_t stream)
{
    const float* x  = (const float*)d_in[0];
    const int*   ei = (const int*)d_in[1];
    // d_in order: x, edge_index, edge_attr, lin_ws[0..2], film_ws[0..2],
    //             film_bs[0..2], skip_ws[0..2], film_skip_ws[0..2]
    const float* lin_w[3]  = {(const float*)d_in[3],  (const float*)d_in[4],  (const float*)d_in[5]};
    const float* film_w[3] = {(const float*)d_in[6],  (const float*)d_in[7],  (const float*)d_in[8]};
    const float* film_b[3] = {(const float*)d_in[9],  (const float*)d_in[10], (const float*)d_in[11]};
    const float* skip_w[3] = {(const float*)d_in[12], (const float*)d_in[13], (const float*)d_in[14]};
    const float* fskip_w[3]= {(const float*)d_in[15], (const float*)d_in[16], (const float*)d_in[17]};

    // ---- workspace layout (all 256B aligned) ----
    char* ws = (char*)d_ws;
    size_t off = 0;
    auto carve = [&](size_t bytes) { void* p = ws + off; off += (bytes + 255) & ~size_t(255); return p; };
    _Float16* xh    = (_Float16*)carve((size_t)N_NODES * 128 * 2); // f16 features
    float*    hbuf  = (float*)   carve((size_t)N_NODES * 128 * 4); // x@lin_w
    float*    bg2   = (float*)   carve((size_t)N_NODES * 256 * 4); // [beta|gamma] (reused)
    float*    agg   = (float*)   carve((size_t)N_NODES * 128 * 4);
    float*    cnt   = (float*)   carve((size_t)N_NODES * 4);
    _Float16* wpack = (_Float16*)carve((size_t)128 * 256 * 2);     // packed weights
    float*    bufA  = (float*)   carve((size_t)N_NODES * 128 * 4);
    float*    bufB  = (float*)   carve((size_t)N_NODES * 128 * 4);
    (void)ws_size; (void)in_sizes; (void)n_in; (void)out_size;

    const int din_a[3]  = {128, 128, 128};
    const int dout_a[3] = {128, 128, 64};
    const float* cur = x;
    float* outs[3] = {bufB, bufA, bufB};

    for (int L = 0; L < 3; ++L) {
        const int din = din_a[L], dout = dout_a[L];
        const int shift = (dout == 128) ? 7 : 6;
        float* B = outs[L];

        // features -> f16
        {
            int n = N_NODES * din;
            cvt_f32_f16<<<(n + 255) / 256, 256, 0, stream>>>(cur, xh, n);
        }
        auto pack = [&](const float* W, int Kd, int Nd) {
            int tot = Kd * Nd;
            pack_w<<<(tot + 255) / 256, 256, 0, stream>>>(W, wpack, Kd, Nd);
        };
        auto gemm0 = [&](float* Cout, int Nd, const float* bias) {
            dim3 g(((Nd >> 4) + 7) / 8, N_NODES / 16);
            gemm_wmma_f16<0><<<g, 256, 0, stream>>>(xh, wpack, Cout, bias, nullptr, Nd, din);
        };

        // 1) self FiLM params: bg_s = x @ film_skip_w  -> bg2 (temporarily)
        pack(fskip_w[L], din, 2 * dout);
        gemm0(bg2, 2 * dout, nullptr);
        // 2) self path fused: B = relu(bg_s[:,dout:]*(x@skip_w) + bg_s[:,:dout])
        pack(skip_w[L], din, dout);
        {
            dim3 g(((dout >> 4) + 7) / 8, N_NODES / 16);
            gemm_wmma_f16<1><<<g, 256, 0, stream>>>(xh, wpack, B, nullptr, bg2, dout, din);
        }
        // 3) edge FiLM params: bg2 = x @ film_w + film_b  (overwrites bg_s)
        pack(film_w[L], din, 2 * dout);
        gemm0(bg2, 2 * dout, film_b[L]);
        // 4) h = x @ lin_w
        pack(lin_w[L], din, dout);
        gemm0(hbuf, dout, nullptr);

        // 5) edge aggregation (atomics, L2-resident)
        hipMemsetAsync(agg, 0, (size_t)N_NODES * dout * 4, stream);
        hipMemsetAsync(cnt, 0, (size_t)N_NODES * 4, stream);
        long long total = (long long)N_EDGES * dout;
        edge_kernel<<<(unsigned)((total + 255) / 256), 256, 0, stream>>>(
            ei, hbuf, bg2, agg, cnt, dout, shift, total);

        // 6) combine + post-layer relu
        int n2 = N_NODES * dout;
        finalize_kernel<<<(n2 + 255) / 256, 256, 0, stream>>>(B, agg, cnt, n2, shift);

        cur = B;
    }

    // log_softmax over last dim (64)
    logsoftmax64<<<(N_NODES + 7) / 8, 256, 0, stream>>>(cur, (float*)d_out, N_NODES);
}
